// _RNN_wrapper_77867757076988
// MI455X (gfx1250) — matrix operand order repaired and checked by reading, not compiled, not run on hardware
//
#include <hip/hip_runtime.h>
#include <math.h>

// Problem constants
#define B_ 8
#define T_ 64
#define L_ 40
#define D_ 512
#define P_ 16
#define H_ 128
#define C_ 22
#define M_ (B_*T_)            // 512 rows (b,t)
#define MTOT (L_*M_)          // 20480 rows for the big GEMM

typedef __attribute__((ext_vector_type(16))) _Float16 v16h;
typedef __attribute__((ext_vector_type(8)))  _Float16 v8h;
typedef __attribute__((ext_vector_type(2)))  _Float16 v2h;
typedef __attribute__((ext_vector_type(8)))  float    v8f;
typedef __attribute__((ext_vector_type(4)))  float    v4f;

union V16 { v16h v; v8h h[2]; };

// ---------------------------------------------------------------------------
// Kernel 1: mean-pool over P=16 and convert to f16.
// features [B,T,L,D,P] f32  ->  X [L, M, D] f16   (X[(l*M+m)*D + d])
// Each thread handles 2 consecutive d (reads 128 contiguous bytes, writes 4B).
// Pure bandwidth: 671 MB read @ 23.3 TB/s dominates the whole problem.
// features is streamed ONCE and is 3.5x the 192MB L2 -> use non-temporal
// loads so it doesn't evict X/XP, which the WMMA kernels re-read from L2.
// ---------------------------------------------------------------------------
__global__ __launch_bounds__(256) void pool_kernel(const float* __restrict__ feat,
                                                   _Float16* __restrict__ X) {
  long tid = (long)blockIdx.x * blockDim.x + threadIdx.x; // total L_*M_*(D_/2)
  int  dd   = (int)(tid & 255);          // pair index: d = dd*2
  long rest = tid >> 8;
  int  m    = (int)(rest & 511);
  int  l    = (int)(rest >> 9);
  if (l >= L_) return;
  long inoff = ((long)m * L_ + l) * (long)(D_*P_) + (long)dd * 32; // 2 d * 16 p
  const v4f* f4 = (const v4f*)(feat + inoff);
  float s0 = 0.f, s1 = 0.f;
#pragma unroll
  for (int i = 0; i < 4; ++i) {
    v4f v = __builtin_nontemporal_load(f4 + i);
    s0 += (v.x + v.y) + (v.z + v.w);
  }
#pragma unroll
  for (int i = 4; i < 8; ++i) {
    v4f v = __builtin_nontemporal_load(f4 + i);
    s1 += (v.x + v.y) + (v.z + v.w);
  }
  v2h o;
  o.x = (_Float16)(s0 * (1.0f/16.0f));
  o.y = (_Float16)(s1 * (1.0f/16.0f));
  *(v2h*)(X + ((long)l*M_ + m)*D_ + (long)dd*2) = o;
}

// ---------------------------------------------------------------------------
// Kernel 2: convert W_ih [H,D] and W_hh [H,H] f32 -> f16 (row-major [n][k]).
// ---------------------------------------------------------------------------
__global__ __launch_bounds__(256) void conv_weights(const float* __restrict__ Wih,
                                                    const float* __restrict__ Whh,
                                                    _Float16* __restrict__ wih_h,
                                                    _Float16* __restrict__ whh_h) {
  int tid = blockIdx.x * 256 + threadIdx.x;
  if (tid < H_*D_) wih_h[tid] = (_Float16)Wih[tid];
  int t2 = tid - H_*D_;
  if (t2 >= 0 && t2 < H_*H_) whh_h[t2] = (_Float16)Whh[t2];
}

// ---------------------------------------------------------------------------
// Kernel 3: hoisted input projection (h-independent part of the RNN step):
//   XP[l,m,h] = sum_d X[l,m,d] * W_ih[h,d] + b_ih[h]
// One [20480 x 512] x [512 x 128] GEMM via v_wmma_f32_16x16x32_f16.
// One output tile (16x16) per wave; 8 waves / block; K-loop of 16 WMMAs.
// ---------------------------------------------------------------------------
__global__ __launch_bounds__(256) void gemm_xproj(const _Float16* __restrict__ X,
                                                  const _Float16* __restrict__ Wih,
                                                  const float* __restrict__ b_ih,
                                                  float* __restrict__ XP) {
  const int wave = threadIdx.x >> 5;
  const int lane = threadIdx.x & 31;
  const int tile = blockIdx.x * 8 + wave;   // 0..(MTOT/16)*(H_/16)-1 = 10239
  const int n0   = (tile & 7) * 16;
  const int M0   = (tile >> 3) * 16;
  const int lr   = lane & 15;
  const int g    = lane >> 4;

  v8f acc = {};
  const _Float16* arow = X   + (long)(M0 + lr) * D_ + g*16;  // A: row=M, K contiguous
  const _Float16* brow = Wih + (long)(n0 + lr) * D_ + g*16;  // B: col=N, K contiguous
#pragma unroll 4
  for (int k0 = 0; k0 < D_; k0 += 32) {
    V16 a, b;
    a.h[0] = *(const v8h*)(arow + k0);        // K = k0 + g*16 + [0..7]
    a.h[1] = *(const v8h*)(arow + k0 + 8);    // K = k0 + g*16 + [8..15]
    b.h[0] = *(const v8h*)(brow + k0);        // K = k0 + g*16 + [0..7]
    b.h[1] = *(const v8h*)(brow + k0 + 8);    // K = k0 + g*16 + [8..15]
    acc = __builtin_amdgcn_wmma_f32_16x16x32_f16(false, a.v, false, b.v,
                                                 (short)0, acc, false, false);
  }
  const float bias = b_ih[n0 + lr];
  float* orow = XP + (long)(M0 + g*8) * H_ + (n0 + lr);
#pragma unroll
  for (int r = 0; r < 8; ++r)
    orow[(long)r * H_] = acc[r] + bias;       // C: elem r -> row M0+g*8+r, col n0+lr
}

// ---------------------------------------------------------------------------
// Kernel 4: the recurrent scan + fused final FC.
// Key insight: the recurrence is independent per row m — only H couples.
// So 32 blocks each own 16 rows; h tile (16x128 f16) ping-pongs in LDS.
// 8 waves/block: wave w owns output columns [w*16, w*16+16); W_hh B-fragments
// live in registers for all 40 steps. 4 WMMAs / wave / step.
// ---------------------------------------------------------------------------
__global__ __launch_bounds__(256) void rnn_scan(const float* __restrict__ XP,
                                                const _Float16* __restrict__ Whh,
                                                const float* __restrict__ b_hh,
                                                const int* __restrict__ len_tubes,
                                                const float* __restrict__ W_fc,
                                                const float* __restrict__ b_fc,
                                                float* __restrict__ out) {
  __shared__ _Float16 sh_h[2][16][H_];   // ping-pong hidden state, 8 KB
  __shared__ int      sh_len[16];

  const int tid    = threadIdx.x;
  const int m_base = blockIdx.x * 16;    // this block's 16 rows

  for (int i = tid; i < 2*16*H_; i += 256) ((_Float16*)sh_h)[i] = (_Float16)0.0f;
  if (tid < 16) sh_len[tid] = len_tubes[m_base + tid];
  __syncthreads();

  const int wave = tid >> 5;
  const int lane = tid & 31;
  const int n0   = wave * 16;
  const int lr   = lane & 15;
  const int g    = lane >> 4;

  // Preload W_hh B-fragments once (reused across all 40 steps).
  V16 bfrag[4];
  {
    const _Float16* brow = Whh + (long)(n0 + lr) * H_ + g*16;
#pragma unroll
    for (int kc = 0; kc < 4; ++kc) {
      bfrag[kc].h[0] = *(const v8h*)(brow + kc*32);
      bfrag[kc].h[1] = *(const v8h*)(brow + kc*32 + 8);
    }
  }
  const float bh = b_hh[n0 + lr];

  for (int l = 0; l < L_; ++l) {
    const _Float16* cur = &sh_h[l & 1][0][0];
    _Float16*       nxt = &sh_h[(l + 1) & 1][0][0];
    const float*    xpl = XP + ((long)l * M_ + m_base) * H_;

    v8f acc = {};
    const _Float16* arow = cur + (long)lr * H_ + g*16;
#pragma unroll
    for (int kc = 0; kc < 4; ++kc) {
      V16 a;
      a.h[0] = *(const v8h*)(arow + kc*32);
      a.h[1] = *(const v8h*)(arow + kc*32 + 8);
      acc = __builtin_amdgcn_wmma_f32_16x16x32_f16(false, a.v, false, bfrag[kc].v,
                                                   (short)0, acc, false, false);
    }
#pragma unroll
    for (int r = 0; r < 8; ++r) {
      const int mloc = g*8 + r;            // local row 0..15
      const int n    = n0 + lr;
      const float v  = acc[r] + xpl[(long)mloc * H_ + n] + bh;
      const _Float16 hn = (_Float16)tanhf(v);
      const bool valid  = l < sh_len[mloc];          // freeze after len
      nxt[mloc * H_ + n] = valid ? hn : cur[mloc * H_ + n];
    }
    __syncthreads();
  }

  // Final h is in buffer (L_ & 1) == 0. Fused FC: logits = h @ W_fc^T + b_fc,
  // zeroed where len == 0.  16 rows x 22 cols = 352 outputs, 256 threads.
  const _Float16* hf = &sh_h[0][0][0];
  for (int idx = tid; idx < 16 * C_; idx += 256) {
    const int mloc = idx / C_;
    const int c    = idx % C_;
    float r = 0.0f;
    if (sh_len[mloc] > 0) {
      const float* wrow = W_fc + (long)c * H_;
      float s = 0.0f;
#pragma unroll 8
      for (int hh = 0; hh < H_; ++hh) s += (float)hf[mloc * H_ + hh] * wrow[hh];
      r = s + b_fc[c];
    }
    out[(long)(m_base + mloc) * C_ + c] = r;
  }
}

// ---------------------------------------------------------------------------
extern "C" void kernel_launch(void* const* d_in, const int* in_sizes, int n_in,
                              void* d_out, int out_size, void* d_ws, size_t ws_size,
                              hipStream_t stream) {
  const float* feat      = (const float*)d_in[0];
  const int*   len_tubes = (const int*)  d_in[1];
  // d_in[2] = target_lbl (unused by forward)
  const float* W_ih      = (const float*)d_in[3];
  const float* W_hh      = (const float*)d_in[4];
  const float* b_ih      = (const float*)d_in[5];
  const float* b_hh      = (const float*)d_in[6];
  const float* W_fc      = (const float*)d_in[7];
  const float* b_fc      = (const float*)d_in[8];
  float* out = (float*)d_out;

  // Workspace layout (bytes, 256-aligned):
  //   X     f16 [MTOT, D]  : 20480*512*2  = 20,971,520
  //   XP    f32 [MTOT, H]  : 20480*128*4  = 10,485,760
  //   wih_h f16 [H, D]     : 131,072
  //   whh_h f16 [H, H]     : 32,768        (total ~31.7 MB)
  char* ws = (char*)d_ws;
  _Float16* X     = (_Float16*)ws;
  float*    XP    = (float*)   (ws + 20971520);
  _Float16* wih_h = (_Float16*)(ws + 31457280);
  _Float16* whh_h = (_Float16*)(ws + 31588352);

  // 1) mean-pool + f32->f16 (bandwidth bound: 671 MB streamed, NT loads)
  pool_kernel<<<(L_*M_*(D_/2))/256, 256, 0, stream>>>(feat, X);
  // 2) weight precision convert (tiny)
  conv_weights<<<(H_*D_ + H_*H_ + 255)/256, 256, 0, stream>>>(W_ih, W_hh, wih_h, whh_h);
  // 3) hoisted input-projection GEMM (WMMA): 10240 tiles / 8 waves per block
  gemm_xproj<<<(MTOT/16)*(H_/16)/8, 256, 0, stream>>>(X, wih_h, b_ih, XP);
  // 4) per-row-group recurrent scan (WMMA) + fused FC head
  rnn_scan<<<M_/16, 256, 0, stream>>>(XP, whh_h, b_hh, len_tubes, W_fc, b_fc, out);
}
